// Generator_31370441130465
// MI455X (gfx1250) — matrix operand order repaired and checked by reading, not compiled, not run on hardware
//
#include <hip/hip_runtime.h>
#include <hip/hip_bf16.h>
#include <math.h>

// ---------------------------------------------------------------------------
// CDNA5 WMMA types (wave32): D(f32 16x16) = A(bf16 16x32) * B(bf16 32x16) + C
// ---------------------------------------------------------------------------
typedef __attribute__((ext_vector_type(16))) __bf16 v16bf;
typedef __attribute__((ext_vector_type(8)))  __bf16 bf16x8;
typedef __attribute__((ext_vector_type(8)))  float  v8f;
typedef int v4i_vs __attribute__((__vector_size__(16)));  // matches builtin proto

#if defined(__has_builtin)
#if __has_builtin(__builtin_amdgcn_global_load_async_to_lds_b128) && \
    __has_builtin(__builtin_amdgcn_s_wait_asynccnt)
#define USE_ASYNC_LDS 1
#endif
#endif
#ifndef USE_ASYNC_LDS
#define USE_ASYNC_LDS 0
#endif

#if USE_ASYNC_LDS
__device__ __forceinline__ void async_cp16(const __bf16* g, __bf16* l) {
  __builtin_amdgcn_global_load_async_to_lds_b128((v4i_vs*)g, (v4i_vs*)l, 0, 0);
}
#endif

__device__ __forceinline__ void sync_cp16(const __bf16* g, __bf16* l) {
  *(bf16x8*)l = *(const bf16x8*)g;
}

// A-fragment (16x32 bf16) ISA layout: lanes 0-15: M=lane, K={0..7,16..23};
// lanes 16-31: M=lane-16, K={8..15,24..31}.  klo = (lane>=16)?8:0 (+ks)
__device__ __forceinline__ v16bf load_frag_a(const __bf16* row, int klo) {
  bf16x8 lo = *(const bf16x8*)(row + klo);
  bf16x8 hi = *(const bf16x8*)(row + klo + 16);
  v16bf r;
#pragma unroll
  for (int e = 0; e < 8; ++e) { r[e] = lo[e]; r[e + 8] = hi[e]; }
  return r;
}

// B-fragment (32x16 bf16): lanes 0-15: N=lane, K=0..15; lanes 16-31: K=16..31
__device__ __forceinline__ v16bf load_frag_b(const __bf16* row, int kb) {
  bf16x8 lo = *(const bf16x8*)(row + kb);
  bf16x8 hi = *(const bf16x8*)(row + kb + 8);
  v16bf r;
#pragma unroll
  for (int e = 0; e < 8; ++e) { r[e] = lo[e]; r[e + 8] = hi[e]; }
  return r;
}

__device__ __forceinline__ v8f wmma_bf16(v16bf a, v16bf b, v8f c) {
  return __builtin_amdgcn_wmma_f32_16x16x32_bf16(false, a, false, b, (short)0, c,
                                                 false, false);
}

// ---------------------------------------------------------------------------
// GEMM: C[M,N] = alpha * A[M,K] @ W^T + bias, A,W bf16 row-major (W is [N,K]).
// Block: 256 threads = 8 waves; tile 128x128, K-tile 64 (2 WMMA substeps),
// double-buffered LDS staged with async-to-LDS when available.
//   TS : per-2048-row-batch transposed store (used to produce V^T)
//   OBF: store bf16 output (else f32)
// ---------------------------------------------------------------------------
template <bool TS, bool OBF>
__global__ __launch_bounds__(256) void gemm_bt(
    const __bf16* __restrict__ A, const __bf16* __restrict__ B,
    const float* __restrict__ bias, void* __restrict__ Cout,
    int M, int N, int K, const float* __restrict__ alphaPtr,
    long long strideA, long long strideB, long long strideC) {
  __shared__ __align__(16) __bf16 sA[2][128][72];  // [m][k], stride 72 keeps 16B align
  __shared__ __align__(16) __bf16 sB[2][128][72];  // [n][k]

  const int tid  = threadIdx.x;
  const int lane = tid & 31;
  const int wave = tid >> 5;
  const int l15  = lane & 15;
  const int half = lane >> 4;

  const int m0 = blockIdx.y * 128;
  const int n0 = blockIdx.x * 128;
  const int z  = blockIdx.z;

  const __bf16* Ab = A + (size_t)z * (size_t)strideA;
  const __bf16* Bb = B + (size_t)z * (size_t)strideB;

  const int wm = (wave & 3) * 32;   // wave M offset: 0/32/64/96
  const int wn = (wave >> 2) * 64;  // wave N offset: 0/64

  v8f acc[2][4];
#pragma unroll
  for (int i = 0; i < 2; ++i)
#pragma unroll
    for (int j = 0; j < 4; ++j)
      acc[i][j] = (v8f){0.f, 0.f, 0.f, 0.f, 0.f, 0.f, 0.f, 0.f};

  // Staging: 128 rows x 64 k per tile; per thread 32 bf16 = 4x16B chunks.
  const int srow = tid >> 1;
  const int skb  = (tid & 1) * 32;
  const __bf16* gA = Ab + (size_t)(m0 + srow) * K + skb;
  const __bf16* gB = Bb + (size_t)(n0 + srow) * K + skb;

  auto stage = [&](int buf, int kt) {
#pragma unroll
    for (int c = 0; c < 4; ++c) {
#if USE_ASYNC_LDS
      async_cp16(gA + kt + c * 8, &sA[buf][srow][skb + c * 8]);
      async_cp16(gB + kt + c * 8, &sB[buf][srow][skb + c * 8]);
#else
      sync_cp16(gA + kt + c * 8, &sA[buf][srow][skb + c * 8]);
      sync_cp16(gB + kt + c * 8, &sB[buf][srow][skb + c * 8]);
#endif
    }
#if !USE_ASYNC_LDS
    if (kt + 64 < K) {
      __builtin_prefetch(gA + kt + 64, 0, 1);
      __builtin_prefetch(gB + kt + 64, 0, 1);
    }
#endif
  };

  stage(0, 0);
  const int nkt = K >> 6;  // K / 64
  for (int it = 0; it < nkt; ++it) {
    const int buf = it & 1;
#if USE_ASYNC_LDS
    __builtin_amdgcn_s_wait_asynccnt(0);
#endif
    __syncthreads();
    if (it + 1 < nkt) stage(buf ^ 1, (it + 1) << 6);

#pragma unroll
    for (int s = 0; s < 2; ++s) {
      const int ks  = s * 32;
      const int klo = ks + half * 8;
      const int kbb = ks + half * 16;
      v16bf af[2], bf[4];
#pragma unroll
      for (int i = 0; i < 2; ++i)
        af[i] = load_frag_a(&sA[buf][wm + i * 16 + l15][0], klo);
#pragma unroll
      for (int j = 0; j < 4; ++j)
        bf[j] = load_frag_a(&sB[buf][wn + j * 16 + l15][0], klo);
#pragma unroll
      for (int i = 0; i < 2; ++i)
#pragma unroll
        for (int j = 0; j < 4; ++j)
          acc[i][j] = wmma_bf16(af[i], bf[j], acc[i][j]);
    }
#if !USE_ASYNC_LDS
    __syncthreads();
#endif
  }

  const float alpha = alphaPtr ? alphaPtr[0] : 1.0f;
  // D layout: VGPR r -> M = r + 8*(lane>=16), N = lane&15
#pragma unroll
  for (int i = 0; i < 2; ++i)
#pragma unroll
    for (int j = 0; j < 4; ++j) {
      const int nG = n0 + wn + j * 16 + l15;
      const float bv = bias ? bias[nG] : 0.0f;
#pragma unroll
      for (int r = 0; r < 8; ++r) {
        const int mG = m0 + wm + i * 16 + r + half * 8;
        const float vl = acc[i][j][r] * alpha + bv;
        size_t idx;
        if (TS)  // per-2048-row batch, store transposed [N][2048]
          idx = ((size_t)(mG >> 11) * N + nG) * 2048 + (size_t)(mG & 2047);
        else
          idx = (size_t)z * (size_t)strideC + (size_t)mG * N + nG;
        if (OBF)
          ((__bf16*)Cout)[idx] = (__bf16)vl;
        else
          ((float*)Cout)[idx] = vl;
      }
    }
}

// ---------------------------------------------------------------------------
// f32 -> bf16 elementwise conversion
// ---------------------------------------------------------------------------
__global__ __launch_bounds__(256) void cvt_bf16_kernel(const float* __restrict__ src,
                                                       __bf16* __restrict__ dst, int n) {
  int i = blockIdx.x * 256 + threadIdx.x;
  if (i < n) dst[i] = (__bf16)src[i];
}

// ---------------------------------------------------------------------------
// Spectral norm: v = W^T u; sigma = u^T W (v/||v||); scale = 1/sigma
// ---------------------------------------------------------------------------
__global__ void sn_v_kernel(const float* __restrict__ W, const float* __restrict__ u,
                            float* __restrict__ v, int N, int K) {
  int k = blockIdx.x * blockDim.x + threadIdx.x;
  if (k < K) {
    float s = 0.f;
    for (int n = 0; n < N; ++n) s += W[(size_t)n * K + k] * u[n];
    v[k] = s;
  }
}

__global__ __launch_bounds__(1024) void sn_sigma_kernel(
    const float* __restrict__ W, const float* __restrict__ u,
    const float* __restrict__ v, float* __restrict__ scale, int N, int K) {
  __shared__ float red[1024];
  __shared__ float invn_s;
  const int t = threadIdx.x;
  float s = 0.f;
  if (t < K) { float x = v[t]; s = x * x; }
  red[t] = s; __syncthreads();
  for (int off = 512; off > 0; off >>= 1) {
    if (t < off) red[t] += red[t + off];
    __syncthreads();
  }
  if (t == 0) invn_s = 1.0f / (sqrtf(red[0]) + 1e-12f);
  __syncthreads();
  const float invn = invn_s;
  float d = 0.f;
  if (t < N) {
    float accv = 0.f;
    for (int k = 0; k < K; ++k) accv += W[(size_t)t * K + k] * v[k];
    d = u[t] * accv * invn;
  }
  __syncthreads();
  red[t] = d; __syncthreads();
  for (int off = 512; off > 0; off >>= 1) {
    if (t < off) red[t] += red[t + off];
    __syncthreads();
  }
  if (t == 0) scale[0] = 1.0f / red[0];
}

// ---------------------------------------------------------------------------
// BatchNorm over axes (0,2) of [B=8, S=2048, L=1024]; channel = seq position s
// ---------------------------------------------------------------------------
__global__ __launch_bounds__(256) void bn_stats_kernel(const float* __restrict__ X,
                                                       float* __restrict__ mean,
                                                       float* __restrict__ var) {
  __shared__ float r1[256], r2[256];
  const int s = blockIdx.x;
  const int t = threadIdx.x;
  float sum = 0.f, sq = 0.f;
  for (int i = t; i < 8192; i += 256) {
    int b = i >> 10, d = i & 1023;
    float x = X[((size_t)(b * 2048 + s) << 10) + d];
    sum += x; sq += x * x;
  }
  r1[t] = sum; r2[t] = sq; __syncthreads();
  for (int off = 128; off > 0; off >>= 1) {
    if (t < off) { r1[t] += r1[t + off]; r2[t] += r2[t + off]; }
    __syncthreads();
  }
  if (t == 0) {
    float m = r1[0] * (1.0f / 8192.0f);
    mean[s] = m;
    var[s] = r2[0] * (1.0f / 8192.0f) - m * m;  // biased variance
  }
}

__global__ __launch_bounds__(256) void bn_apply_lrelu_kernel(
    const float* __restrict__ X, __bf16* __restrict__ Y,
    const float* __restrict__ mean, const float* __restrict__ var,
    const float* __restrict__ g, const float* __restrict__ be) {
  size_t idx = (size_t)blockIdx.x * 256 + threadIdx.x;  // < 16777216
  int s = (int)((idx >> 10) & 2047);
  float x = X[idx];
  float xn = (x - mean[s]) * rsqrtf(var[s] + 1e-5f) * g[s] + be[s];
  Y[idx] = (__bf16)(xn > 0.f ? xn : 0.2f * xn);
}

// ---------------------------------------------------------------------------
// Row softmax (rows of 2048 f32), writing bf16 for the following GEMM
// ---------------------------------------------------------------------------
__global__ __launch_bounds__(256) void softmax_rows_kernel(float* __restrict__ S,
                                                           __bf16* __restrict__ Sb) {
  __shared__ float red[256];
  const int t = threadIdx.x;
  float* p = S + ((size_t)blockIdx.x << 11);
  __bf16* pb = Sb + ((size_t)blockIdx.x << 11);
  float m = -3.402823e38f;
  for (int i = t; i < 2048; i += 256) m = fmaxf(m, p[i]);
  red[t] = m; __syncthreads();
  for (int off = 128; off > 0; off >>= 1) {
    if (t < off) red[t] = fmaxf(red[t], red[t + off]);
    __syncthreads();
  }
  const float mx = red[0]; __syncthreads();
  float sum = 0.f;
  for (int i = t; i < 2048; i += 256) {
    float e = __expf(p[i] - mx);
    p[i] = e; sum += e;
  }
  red[t] = sum; __syncthreads();
  for (int off = 128; off > 0; off >>= 1) {
    if (t < off) red[t] += red[t + off];
    __syncthreads();
  }
  const float inv = 1.0f / red[0]; __syncthreads();
  for (int i = t; i < 2048; i += 256) pb[i] = (__bf16)(p[i] * inv);
}

// ---------------------------------------------------------------------------
// Host-side orchestration
// ---------------------------------------------------------------------------
static inline void run_gemm(int ts, int obf, const __bf16* A, const __bf16* B,
                            const float* bias, void* C, int M, int N, int K,
                            const float* alpha, long long sA, long long sB,
                            long long sC, int batch, hipStream_t st) {
  dim3 grid(N / 128, M / 128, batch), blk(256);
  if (ts)
    gemm_bt<true, true><<<grid, blk, 0, st>>>(A, B, bias, C, M, N, K, alpha, sA, sB, sC);
  else if (obf)
    gemm_bt<false, true><<<grid, blk, 0, st>>>(A, B, bias, C, M, N, K, alpha, sA, sB, sC);
  else
    gemm_bt<false, false><<<grid, blk, 0, st>>>(A, B, bias, C, M, N, K, alpha, sA, sB, sC);
}

static inline void run_cvt(const float* s, __bf16* d, int n, hipStream_t st) {
  cvt_bf16_kernel<<<(n + 255) / 256, 256, 0, st>>>(s, d, n);
}

extern "C" void kernel_launch(void* const* d_in, const int* in_sizes, int n_in,
                              void* d_out, int out_size, void* d_ws, size_t ws_size,
                              hipStream_t stream) {
  const float* x   = (const float*)d_in[0];
  const float* W1  = (const float*)d_in[1];
  const float* b1  = (const float*)d_in[2];
  const float* u1  = (const float*)d_in[3];
  const float* g1  = (const float*)d_in[4];
  const float* be1 = (const float*)d_in[5];
  const float* Wq1 = (const float*)d_in[6];
  const float* bq1 = (const float*)d_in[7];
  const float* Wk1 = (const float*)d_in[8];
  const float* bk1 = (const float*)d_in[9];
  const float* Wv1 = (const float*)d_in[10];
  const float* bv1 = (const float*)d_in[11];
  const float* W2  = (const float*)d_in[12];
  const float* b2  = (const float*)d_in[13];
  const float* u2  = (const float*)d_in[14];
  const float* g2  = (const float*)d_in[15];
  const float* be2 = (const float*)d_in[16];
  const float* Wq2 = (const float*)d_in[17];
  const float* bq2 = (const float*)d_in[18];
  const float* Wk2 = (const float*)d_in[19];
  const float* bk2 = (const float*)d_in[20];
  const float* Wv2 = (const float*)d_in[21];
  const float* bv2 = (const float*)d_in[22];
  const float* W3  = (const float*)d_in[23];
  const float* b3  = (const float*)d_in[24];
  const float* u3  = (const float*)d_in[25];
  float* out = (float*)d_out;

  const long long MB = 1024LL * 1024LL;
  char* ws = (char*)d_ws;
  __bf16* xb  = (__bf16*)(ws + 0 * MB);     // [16384,512]
  __bf16* Hb  = (__bf16*)(ws + 16 * MB);    // [16384,1024] bf16 activations
  __bf16* Qb  = (__bf16*)(ws + 48 * MB);
  __bf16* Kb  = (__bf16*)(ws + 80 * MB);
  __bf16* Vtb = (__bf16*)(ws + 112 * MB);   // [8][1024][2048] V transposed
  float*  SCf = (float*)(ws + 144 * MB);    // [8,2048,2048] f32 scores
  __bf16* SCb = (__bf16*)(ws + 272 * MB);   // bf16 softmaxed scores
  float*  Tf  = (float*)(ws + 336 * MB);    // [16384,1024] f32 pre-BN
  __bf16* Ab  = (__bf16*)(ws + 400 * MB);   // [16384,1024] attn output bf16
  __bf16* wp  = (__bf16*)(ws + 432 * MB);   // bf16 weight pool (16 MB)
  float*  vvec = (float*)(ws + 456 * MB);   // 2048 f32
  float*  scl  = vvec + 2048;               // 3 scalars
  float*  mean = scl + 16;                  // 2048 f32
  float*  var  = mean + 2048;               // 2048 f32

  __bf16* W1b  = wp;               // 1024x512
  __bf16* Wq1b = W1b + 524288;
  __bf16* Wk1b = Wq1b + 1048576;
  __bf16* Wv1b = Wk1b + 1048576;
  __bf16* W2b  = Wv1b + 1048576;
  __bf16* Wq2b = W2b + 1048576;
  __bf16* Wk2b = Wq2b + 1048576;
  __bf16* Wv2b = Wk2b + 1048576;
  __bf16* W3b  = Wv2b + 1048576;   // 512x1024

  const int Mrows = 16384;  // B*S
  const long long sQK = 2048LL * 1024LL;
  const long long sSC = 2048LL * 2048LL;
  const long long sVt = 1024LL * 2048LL;

  // ---- one-time bf16 conversions ----
  run_cvt(x, xb, 16384 * 512, stream);
  run_cvt(W1, W1b, 1024 * 512, stream);
  run_cvt(Wq1, Wq1b, 1024 * 1024, stream);
  run_cvt(Wk1, Wk1b, 1024 * 1024, stream);
  run_cvt(Wv1, Wv1b, 1024 * 1024, stream);
  run_cvt(W2, W2b, 1024 * 1024, stream);
  run_cvt(Wq2, Wq2b, 1024 * 1024, stream);
  run_cvt(Wk2, Wk2b, 1024 * 1024, stream);
  run_cvt(Wv2, Wv2b, 1024 * 1024, stream);
  run_cvt(W3, W3b, 512 * 1024, stream);

  // ---- spectral-norm scales (f32 weights; folded into GEMM alpha) ----
  sn_v_kernel<<<2, 256, 0, stream>>>(W1, u1, vvec, 1024, 512);
  sn_sigma_kernel<<<1, 1024, 0, stream>>>(W1, u1, vvec, scl + 0, 1024, 512);
  sn_v_kernel<<<4, 256, 0, stream>>>(W2, u2, vvec, 1024, 1024);
  sn_sigma_kernel<<<1, 1024, 0, stream>>>(W2, u2, vvec, scl + 1, 1024, 1024);
  sn_v_kernel<<<4, 256, 0, stream>>>(W3, u3, vvec, 512, 1024);
  sn_sigma_kernel<<<1, 1024, 0, stream>>>(W3, u3, vvec, scl + 2, 512, 1024);

  // ---- layer 1: lin -> BN -> leaky-relu ----
  run_gemm(0, 0, xb, W1b, b1, Tf, Mrows, 1024, 512, scl + 0, 0, 0, 0, 1, stream);
  bn_stats_kernel<<<2048, 256, 0, stream>>>(Tf, mean, var);
  bn_apply_lrelu_kernel<<<65536, 256, 0, stream>>>(Tf, Hb, mean, var, g1, be1);

  // ---- attention 1 ----
  run_gemm(0, 1, Hb, Wq1b, bq1, Qb, Mrows, 1024, 1024, nullptr, 0, 0, 0, 1, stream);
  run_gemm(0, 1, Hb, Wk1b, bk1, Kb, Mrows, 1024, 1024, nullptr, 0, 0, 0, 1, stream);
  run_gemm(1, 1, Hb, Wv1b, bv1, Vtb, Mrows, 1024, 1024, nullptr, 0, 0, 0, 1, stream);
  run_gemm(0, 0, Qb, Kb, nullptr, SCf, 2048, 2048, 1024, nullptr, sQK, sQK, sSC, 8, stream);
  softmax_rows_kernel<<<16384, 256, 0, stream>>>(SCf, SCb);
  run_gemm(0, 1, SCb, Vtb, nullptr, Ab, 2048, 1024, 2048, nullptr, sSC, sVt, sQK, 8, stream);

  // ---- layer 2: lin -> BN -> leaky-relu ----
  run_gemm(0, 0, Ab, W2b, b2, Tf, Mrows, 1024, 1024, scl + 1, 0, 0, 0, 1, stream);
  bn_stats_kernel<<<2048, 256, 0, stream>>>(Tf, mean, var);
  bn_apply_lrelu_kernel<<<65536, 256, 0, stream>>>(Tf, Hb, mean, var, g2, be2);

  // ---- attention 2 ----
  run_gemm(0, 1, Hb, Wq2b, bq2, Qb, Mrows, 1024, 1024, nullptr, 0, 0, 0, 1, stream);
  run_gemm(0, 1, Hb, Wk2b, bk2, Kb, Mrows, 1024, 1024, nullptr, 0, 0, 0, 1, stream);
  run_gemm(1, 1, Hb, Wv2b, bv2, Vtb, Mrows, 1024, 1024, nullptr, 0, 0, 0, 1, stream);
  run_gemm(0, 0, Qb, Kb, nullptr, SCf, 2048, 2048, 1024, nullptr, sQK, sQK, sSC, 8, stream);
  softmax_rows_kernel<<<16384, 256, 0, stream>>>(SCf, SCb);
  run_gemm(0, 1, SCb, Vtb, nullptr, Ab, 2048, 1024, 2048, nullptr, sSC, sVt, sQK, 8, stream);

  // ---- final linear D2 -> D straight into d_out ----
  run_gemm(0, 0, Ab, W3b, b3, out, Mrows, 512, 1024, scl + 2, 0, 0, 0, 1, stream);
}
